// StripSMambaNet_82085414961667
// MI455X (gfx1250) — compile-verified, matrix operand order checked by reading, not run on hardware
//
#include <hip/hip_runtime.h>

typedef _Float16 v16h __attribute__((ext_vector_type(16)));
typedef _Float16 h8   __attribute__((ext_vector_type(8)));
typedef float    v8f  __attribute__((ext_vector_type(8)));

#define B_   4
#define C_   96
#define H_   48
#define W_   48
#define HW_  (H_*W_)
#define L_   HW_
#define BL_  (B_*L_)
#define DI_  192
#define DSN_ 16
#define DTR_ 6
#define DC_  4
#define NPROJ_ 48   /* 38 padded to 48 for WMMA tiles */
#define DI2_ 384

__device__ __forceinline__ float sigmoidf_(float x){ return 1.f/(1.f+__expf(-x)); }
__device__ __forceinline__ float siluf_(float x){ return x*sigmoidf_(x); }

__device__ __forceinline__ int src_spatial_(int d, int l, const int* __restrict__ idx){
  if(d==0) return l;
  if(d==1){ int w=l/H_, h=l%H_; return h*W_+w; }
  int p = idx[l];
  if(d==2) return p;
  int i=p/W_, j=p%W_;
  return i*W_ + (W_-1-j);
}

__global__ void k_zero(float* __restrict__ p, int n){
  int i=blockIdx.x*blockDim.x+threadIdx.x; if(i<n) p[i]=0.f;
}

// diag perm: rank sorted by (j-i) primary, i secondary
__global__ void k_diag_perm(int* __restrict__ idx, int* __restrict__ inv){
  int cell = blockIdx.x*blockDim.x+threadIdx.x;
  if(cell>=HW_) return;
  int i = cell / W_, j = cell % W_;
  int k = j - i;
  int off = 0;
  for(int kk=-(H_-1); kk<k; ++kk) off += H_ - (kk<0?-kk:kk);
  int pos = i - (k<0 ? -k : 0);
  int r = off + pos;
  inv[cell] = r;
  idx[r] = cell;
}

__global__ void k_cvt_f16(const float* __restrict__ src, _Float16* __restrict__ dst, int n){
  int i=blockIdx.x*blockDim.x+threadIdx.x;
  if(i<n) dst[i]=(_Float16)src[i];
}

// Wx (4,38,192) -> Wx16 (4,48,192) zero-padded rows 38..47
__global__ void k_cvt_wx_pad(const float* __restrict__ Wx, _Float16* __restrict__ Wx16){
  int i = blockIdx.x*blockDim.x+threadIdx.x;
  if(i >= 4*NPROJ_*DI_) return;
  int kcol = i % DI_;
  int e = (i / DI_) % NPROJ_;
  int d = i / (DI_*NPROJ_);
  Wx16[i] = (e < (DTR_+2*DSN_)) ? (_Float16)Wx[((size_t)d*(DTR_+2*DSN_)+e)*DI_ + kcol]
                                : (_Float16)0.f;
}

// LayerNorm over C=96, directional gather, f16 convert. One wave per token.
__global__ void k_ln_gather(const float* __restrict__ x, const float* __restrict__ ln_g,
                            const float* __restrict__ ln_b, const int* __restrict__ idx,
                            _Float16* __restrict__ t16, int d){
  int lane = threadIdx.x & 31;
  int wave = threadIdx.x >> 5;
  int tok  = blockIdx.x * (blockDim.x>>5) + wave;
  if(tok >= BL_) return;
  int b = tok / L_;
  int l = tok % L_;
  int s = src_spatial_(d, l, idx);
  const float* xb = x + ((size_t)b*C_)*HW_ + s;
  float v0 = xb[(size_t)(lane     )*HW_];
  float v1 = xb[(size_t)(lane + 32)*HW_];
  float v2 = xb[(size_t)(lane + 64)*HW_];
  float sum = v0+v1+v2, sq = v0*v0+v1*v1+v2*v2;
  #pragma unroll
  for(int o=16;o>=1;o>>=1){ sum += __shfl_xor(sum,o,32); sq += __shfl_xor(sq,o,32); }
  float mu  = sum * (1.f/C_);
  float var = sq  * (1.f/C_) - mu*mu;
  float rs  = rsqrtf(var + 1e-5f);
  _Float16* trow = t16 + (size_t)tok*C_;
  const float* g  = ln_g + d*C_;
  const float* be = ln_b + d*C_;
  trow[lane]    = (_Float16)((v0-mu)*rs*g[lane]    + be[lane]);
  trow[lane+32] = (_Float16)((v1-mu)*rs*g[lane+32] + be[lane+32]);
  trow[lane+64] = (_Float16)((v2-mu)*rs*g[lane+64] + be[lane+64]);
}

// NT GEMM: C[M,N] = A[M,K] (f16 row-major) * Wt[N,K] (f16 row-major), f32 out.
// One wave per 16x16 tile, v_wmma_f32_16x16x32_f16 over K/32 steps.
__global__ void __launch_bounds__(128) gemm_nt_f16(
    const _Float16* __restrict__ A, const _Float16* __restrict__ Wt,
    float* __restrict__ C, int M, int N, int K){
  int lane = threadIdx.x & 31;
  int wave = threadIdx.x >> 5;
  int tn = blockIdx.x * 4 + wave;
  int tm = blockIdx.y;
  if (tn * 16 >= N) return;            // wave-uniform guard; EXEC stays all-ones
  int mr = tm*16 + (lane & 15);
  int nr = tn*16 + (lane & 15);
  int ksel = (lane >> 4) * 8;          // lanes 0-15: K 0..7 / 16..23; lanes 16-31: 8..15 / 24..31
  const _Float16* arow = A  + (size_t)mr * K + ksel;
  const _Float16* brow = Wt + (size_t)nr * K + ksel;
  __builtin_prefetch(brow, 0, 3);      // global_prefetch_b8: pull weight row toward WGP
  __builtin_prefetch(arow, 0, 3);
  v8f acc = {};
  for (int k0 = 0; k0 < K; k0 += 32) {
    h8 alo = *(const h8*)(arow + k0);
    h8 ahi = *(const h8*)(arow + k0 + 16);
    h8 blo = *(const h8*)(brow + k0);
    h8 bhi = *(const h8*)(brow + k0 + 16);
    v16h av, bv;
    #pragma unroll
    for (int i = 0; i < 8; ++i) { av[i]=alo[i]; av[i+8]=ahi[i]; bv[i]=blo[i]; bv[i+8]=bhi[i]; }
    acc = __builtin_amdgcn_wmma_f32_16x16x32_f16(false, av, false, bv, (short)0, acc, false, false);
  }
  int mrow = tm*16 + (lane >> 4)*8;    // D layout: VGPR r -> M=r (+8 for upper lanes)
  int ncol = tn*16 + (lane & 15);
  #pragma unroll
  for (int r = 0; r < 8; ++r)
    C[(size_t)(mrow + r) * N + ncol] = acc[r];
}

// causal depthwise conv + SiLU; also SiLU(z). xz layout [tok][384].
__global__ void k_conv_silu(const float* __restrict__ xz, const float* __restrict__ conv_w,
                            const float* __restrict__ conv_b, float* __restrict__ u32,
                            _Float16* __restrict__ u16, float* __restrict__ sz, int d){
  int gid = blockIdx.x*blockDim.x+threadIdx.x;
  if(gid >= BL_*DI_) return;
  int ch  = gid % DI_;
  int tok = gid / DI_;
  int l   = tok % L_;
  float s = conv_b[d*DI_ + ch];
  const float* w = conv_w + ((size_t)d*DI_ + ch)*DC_;
  #pragma unroll
  for(int j=0;j<DC_;++j){
    int ls = l - (DC_-1) + j;
    if(ls >= 0) s += w[j] * xz[(size_t)(tok - (DC_-1) + j)*DI2_ + ch];
  }
  float u = siluf_(s);
  u32[gid] = u;
  u16[gid] = (_Float16)u;
  float z = xz[(size_t)tok*DI2_ + DI_ + ch];
  sz[gid] = siluf_(z);
}

// dt = softplus(dt_r * Wdt^T + bdt), K=6 -> plain VALU.
__global__ void k_dt(const float* __restrict__ proj, const float* __restrict__ Wdt,
                     const float* __restrict__ bdt, float* __restrict__ dtb, int d){
  int gid = blockIdx.x*blockDim.x+threadIdx.x;
  if(gid >= BL_*DI_) return;
  int ch  = gid % DI_;
  int tok = gid / DI_;
  float a = bdt[d*DI_ + ch];
  const float* wr = Wdt + ((size_t)d*DI_ + ch)*DTR_;
  const float* pr = proj + (size_t)tok*NPROJ_;
  #pragma unroll
  for(int r=0;r<DTR_;++r) a += pr[r]*wr[r];
  dtb[gid] = (a > 20.f) ? a : log1pf(__expf(a));
}

// selective scan, state-parallel: one lane per (channel, state).
// Wave handles 2 channels x 16 states; per step: 1 exp + ~6 VALU per lane,
// B_t/C_t are single coalesced 16-lane loads, y-reduction = 4 shfl_xor.
// Folds D skip-connection and SiLU(z) gate; writes f16 for GEMM3.
__global__ void __launch_bounds__(32) k_scan(
    const float* __restrict__ dtb, const float* __restrict__ u32,
    const float* __restrict__ proj, const float* __restrict__ sz,
    const float* __restrict__ A_log, const float* __restrict__ Dp,
    _Float16* __restrict__ y16, int d){
  int lane = threadIdx.x;
  int s    = lane & 15;          // state index
  int sub  = lane >> 4;          // which of the 2 channels in this wave
  int b    = blockIdx.x / (DI_/2);
  int cp   = blockIdx.x % (DI_/2);
  int ch   = cp*2 + sub;
  float Ar = -__expf(A_log[((size_t)d*DI_+ch)*DSN_ + s]);
  float Dv = Dp[d*DI_ + ch];
  float h  = 0.f;
  for(int l=0;l<L_;++l){
    size_t tok = (size_t)b*L_ + l;
    float dtv = dtb[tok*DI_ + ch];             // uniform within 16-lane half
    float uv  = u32[tok*DI_ + ch];
    float szv = sz [tok*DI_ + ch];
    const float* pr = proj + tok*NPROJ_;
    float Bs = pr[DTR_ + s];                   // coalesced 16-lane load
    float Cs = pr[DTR_ + DSN_ + s];
    float dA = __expf(dtv * Ar);
    h = dA*h + (dtv*Bs)*uv;
    float p = h*Cs;
    #pragma unroll
    for(int o=1;o<16;o<<=1) p += __shfl_xor(p, o, 32);   // sum over 16 states
    if(s == 0)
      y16[tok*DI_ + ch] = (_Float16)((p + uv*Dv) * szv);
  }
}

// scatter-accumulate direction feature back to spatial layout (bijective per dir)
__global__ void k_accum(const float* __restrict__ feat, const int* __restrict__ idx,
                        float* __restrict__ acc, int d){
  int gid = blockIdx.x*blockDim.x+threadIdx.x;
  if(gid >= BL_*C_) return;
  int c   = gid % C_;
  int tok = gid / C_;
  int b = tok / L_, l = tok % L_;
  int s = src_spatial_(d, l, idx);
  acc[((size_t)b*C_ + c)*HW_ + s] += feat[(size_t)tok*C_ + c];
}

__global__ void k_gate(const float* __restrict__ x, const float* __restrict__ acc,
                       float* __restrict__ out, int n){
  int i=blockIdx.x*blockDim.x+threadIdx.x;
  if(i<n) out[i] = x[i]*sigmoidf_(acc[i]*0.25f);
}

extern "C" void kernel_launch(void* const* d_in, const int* in_sizes, int n_in,
                              void* d_out, int out_size, void* d_ws, size_t ws_size,
                              hipStream_t stream){
  (void)in_sizes; (void)n_in; (void)out_size; (void)ws_size;
  const float* x      = (const float*)d_in[0];
  const float* ln_g   = (const float*)d_in[1];
  const float* ln_b   = (const float*)d_in[2];
  const float* Win    = (const float*)d_in[3];
  const float* conv_w = (const float*)d_in[4];
  const float* conv_b = (const float*)d_in[5];
  const float* Wx     = (const float*)d_in[6];
  const float* Wdt    = (const float*)d_in[7];
  const float* bdt    = (const float*)d_in[8];
  const float* A_log  = (const float*)d_in[9];
  const float* Dp     = (const float*)d_in[10];
  const float* Wout   = (const float*)d_in[11];
  float* out = (float*)d_out;
  char* ws = (char*)d_ws;

  // 256B-aligned scratch slabs (~53.6 MB total)
  int*      idx    = (int*)     (ws + 0);         // 2304 * 4
  int*      inv    = (int*)     (ws + 9216);      // 2304 * 4 (computed, kept for clarity)
  _Float16* Win16  = (_Float16*)(ws + 18432);     // 4*384*96 f16
  _Float16* Wx16   = (_Float16*)(ws + 313344);    // 4*48*192 f16 (padded)
  _Float16* Wout16 = (_Float16*)(ws + 387072);    // 4*96*192 f16
  _Float16* t16    = (_Float16*)(ws + 534528);    // 9216*96 f16
  float*    xz     = (float*)   (ws + 2304000);   // 9216*384 f32
  float*    u32    = (float*)   (ws + 16459776);  // 9216*192 f32
  _Float16* u16    = (_Float16*)(ws + 23537664);  // 9216*192 f16
  float*    sz     = (float*)   (ws + 27076608);  // 9216*192 f32
  float*    proj   = (float*)   (ws + 34154496);  // 9216*48 f32
  float*    dtb    = (float*)   (ws + 35923968);  // 9216*192 f32
  _Float16* y16    = (_Float16*)(ws + 43001856);  // 9216*192 f16
  float*    feat   = (float*)   (ws + 46540800);  // 9216*96 f32
  float*    acc    = (float*)   (ws + 50079744);  // 4*96*2304 f32
  (void)inv;

  const int TB = 256;
  k_zero<<<(BL_*C_+TB-1)/TB, TB, 0, stream>>>(acc, BL_*C_);
  k_diag_perm<<<(HW_+TB-1)/TB, TB, 0, stream>>>(idx, inv);
  k_cvt_f16<<<(4*DI2_*C_+TB-1)/TB, TB, 0, stream>>>(Win, Win16, 4*DI2_*C_);
  k_cvt_wx_pad<<<(4*NPROJ_*DI_+TB-1)/TB, TB, 0, stream>>>(Wx, Wx16);
  k_cvt_f16<<<(4*C_*DI_+TB-1)/TB, TB, 0, stream>>>(Wout, Wout16, 4*C_*DI_);

  for(int d=0; d<4; ++d){
    k_ln_gather<<<BL_/8, 256, 0, stream>>>(x, ln_g, ln_b, idx, t16, d);
    // GEMM1: xz[9216,384] = t16[9216,96] x Win[d][384,96]^T   (Ntiles=24)
    gemm_nt_f16<<<dim3(6, BL_/16), 128, 0, stream>>>(t16, Win16 + (size_t)d*DI2_*C_, xz, BL_, DI2_, C_);
    k_conv_silu<<<(BL_*DI_+TB-1)/TB, TB, 0, stream>>>(xz, conv_w, conv_b, u32, u16, sz, d);
    // GEMM2: proj[9216,48] = u16[9216,192] x Wx[d][48,192]^T  (Ntiles=3, wave 3 idles)
    gemm_nt_f16<<<dim3(1, BL_/16), 128, 0, stream>>>(u16, Wx16 + (size_t)d*NPROJ_*DI_, proj, BL_, NPROJ_, DI_);
    k_dt<<<(BL_*DI_+TB-1)/TB, TB, 0, stream>>>(proj, Wdt, bdt, dtb, d);
    // state-parallel scan: 2 channels per wave, 96 waves per batch image
    k_scan<<<B_*(DI_/2), 32, 0, stream>>>(dtb, u32, proj, sz, A_log, Dp, y16, d);
    // GEMM3: feat[9216,96] = y16[9216,192] x Wout[d][96,192]^T (Ntiles=6)
    gemm_nt_f16<<<dim3(2, BL_/16), 128, 0, stream>>>(y16, Wout16 + (size_t)d*C_*DI_, feat, BL_, C_, DI_);
    k_accum<<<(BL_*C_+TB-1)/TB, TB, 0, stream>>>(feat, idx, acc, d);
  }
  k_gate<<<(B_*C_*HW_+TB-1)/TB, TB, 0, stream>>>(x, acc, out, B_*C_*HW_);
}
